// IOEncoder_3959959846975
// MI455X (gfx1250) — matrix operand order repaired and checked by reading, not compiled, not run on hardware
//
#include <hip/hip_runtime.h>

// MI455X / gfx1250 attention-LSTM decoder.
// - All GEMMs on v_wmma_f32_16x16x32_f16 (fp32 accumulate), weights f16.
// - Weights (~7MB f16) stay L2-resident across the 48 sequential steps.
// - enc_feat stored f16 so attn_out(100MB f32)+enc_feat(50MB f16) < 192MB L2.
// - LSTM nonlinearity fused into the gates-GEMM epilogue (i/f/g/o tiles per block).
// - Double-buffered LDS tiles, 16B (b128) staging both directions, 1 barrier/iter.

#define HIDN 512
#define EMBD 128
#define BTOT 1024
#define LSEQ 48
#define TKK  48
#define KCAT 1152   // [x(128) | ctx(512) | h(512)]
#define LDT  40     // padded LDS k-stride (halves): 80B rows, 16B aligned

typedef __attribute__((ext_vector_type(16))) _Float16 v16h;
typedef __attribute__((ext_vector_type(8)))  _Float16 v8h;
typedef __attribute__((ext_vector_type(8)))  float    v8f;

__device__ __forceinline__ float sigf(float x){ return 1.0f/(1.0f + __expf(-x)); }

__device__ __forceinline__ v8h cvt8(float4 f0, float4 f1){
  v8h h;
  h[0]=(_Float16)f0.x; h[1]=(_Float16)f0.y; h[2]=(_Float16)f0.z; h[3]=(_Float16)f0.w;
  h[4]=(_Float16)f1.x; h[5]=(_Float16)f1.y; h[6]=(_Float16)f1.z; h[7]=(_Float16)f1.w;
  return h;
}

// A fragment (16x32 f16, MxK), ISA 7.12.2 layout: lane&15 = M row,
// koff=(lane>>4)*8; VGPR0-3 = K koff..+7, VGPR4-7 = K koff+16..+23.
__device__ __forceinline__ v16h frag_A(const _Float16* base, int lane, int m0){
  const _Float16* p = base + (m0 + (lane & 15)) * LDT + ((lane >> 4) << 3);
  v8h lo = *(const v8h*)(p);
  v8h hi = *(const v8h*)(p + 16);
  v16h r;
#pragma unroll
  for (int i = 0; i < 8; ++i){ r[i] = lo[i]; r[i+8] = hi[i]; }
  return r;
}

// B fragment (32x16 f16, KxN) from LDS tile stored [N][K]:
// lane&15 = N col, koff=(lane>>4)*8; VGPR0-3 = K koff..+7, VGPR4-7 = K koff+16..+23.
__device__ __forceinline__ v16h frag_B(const _Float16* base, int lane, int n0){
  const _Float16* p = base + (n0 + (lane & 15)) * LDT + ((lane >> 4) << 3);
  v8h lo = *(const v8h*)(p);
  v8h hi = *(const v8h*)(p + 16);
  v16h r;
#pragma unroll
  for (int i = 0; i < 8; ++i){ r[i] = lo[i]; r[i+8] = hi[i]; }
  return r;
}

#define WMMA(a,b,c) __builtin_amdgcn_wmma_f32_16x16x32_f16(false,(a),false,(b),(short)0,(c),false,false)

// ---------------- prep kernels ----------------
__global__ void k_cvt(const float* __restrict__ s, _Float16* __restrict__ d, int n){
  int i = blockIdx.x*256 + threadIdx.x;
  if (i < n) d[i] = (_Float16)s[i];
}

__global__ void k_prep_wcat(const float* __restrict__ Wih, const float* __restrict__ Whh,
                            _Float16* __restrict__ W){
  int i = blockIdx.x*256 + threadIdx.x;
  if (i >= 2048*KCAT) return;
  int n = i / KCAT, k = i - n*KCAT;
  float v = (k < EMBD + HIDN) ? Wih[n*(EMBD+HIDN) + k] : Whh[n*HIDN + (k - EMBD - HIDN)];
  W[i] = (_Float16)v;
}

__global__ void k_init(const float* __restrict__ h0, const float* __restrict__ c0,
                       float* __restrict__ hout, float* __restrict__ cout,
                       _Float16* __restrict__ h16, _Float16* __restrict__ ctx16){
  int i = blockIdx.x*256 + threadIdx.x;
  if (i >= BTOT*HIDN) return;
  hout[i] = h0[i]; cout[i] = c0[i];
  h16[i]  = (_Float16)h0[i];
  ctx16[i] = (_Float16)0.0f;
}

// ---------------- enc_feat = attn_out @ Wh^T  (49152 x 512, K=512) ----------
__global__ void __launch_bounds__(256) k_encfeat(const float* __restrict__ attn,
                                                 const _Float16* __restrict__ Wh16,
                                                 _Float16* __restrict__ enc16){
  __shared__ __align__(16) _Float16 As[2][64*LDT];
  __shared__ __align__(16) _Float16 Bs[2][64*LDT];
  const int tid = threadIdx.x, lane = tid & 31, w = tid >> 5;
  const int nt = blockIdx.x, mt = blockIdx.y;
  const int sm = (w & 3)*16, sn = (w >> 2)*32;
  const int m = tid >> 2, kc = (tid & 3) * 8;   // one 16B chunk per thread

  auto stage = [&](int buf, int k0){
    const float4* sa = (const float4*)(attn + (size_t)(mt*64 + m)*HIDN + k0 + kc);
    *(v8h*)&As[buf][m*LDT + kc] = cvt8(sa[0], sa[1]);
    *(v8h*)&Bs[buf][m*LDT + kc] =
        *(const v8h*)(Wh16 + (size_t)(nt*64 + m)*HIDN + k0 + kc);
  };

  v8f acc0 = {}, acc1 = {};
  stage(0, 0);
  const int NIT = HIDN/32;
  for (int kk = 0; kk < NIT; ++kk){
    __syncthreads();
    const int cur = kk & 1;
    if (kk + 1 < NIT) stage(cur ^ 1, (kk + 1)*32);
    v16h a  = frag_A(As[cur], lane, sm);
    v16h b0 = frag_B(Bs[cur], lane, sn);
    v16h b1 = frag_B(Bs[cur], lane, sn + 16);
    acc0 = WMMA(a, b0, acc0);
    acc1 = WMMA(a, b1, acc1);
  }
  const int rb = mt*64 + sm + ((lane >> 4) << 3);
  const int cn = nt*64 + sn + (lane & 15);
#pragma unroll
  for (int r = 0; r < 8; ++r){
    enc16[(size_t)(rb + r)*HIDN + cn]      = (_Float16)acc0[r];
    enc16[(size_t)(rb + r)*HIDN + cn + 16] = (_Float16)acc1[r];
  }
}

// ------- fused gates GEMM + LSTM cell: block = 64 rows x 64 hidden units,
// i/f/g/o tiles computed together (K=1152), cell update in epilogue ----------
__global__ void __launch_bounds__(256) k_lstm(
    const int* __restrict__ seq, const int* __restrict__ lens,
    const float* __restrict__ emb, const _Float16* __restrict__ Wcat,
    const _Float16* __restrict__ ctx16,
    const _Float16* __restrict__ h16r, _Float16* __restrict__ h16w,
    const float* __restrict__ bih, const float* __restrict__ bhh,
    float* __restrict__ cout, float* __restrict__ hout, float* __restrict__ out,
    _Float16* __restrict__ hc16, int t)
{
  __shared__ __align__(16) _Float16 As[2][64*LDT];
  __shared__ __align__(16) _Float16 Bs[2][256*LDT];
  const int tid = threadIdx.x, lane = tid & 31, w = tid >> 5;
  const int ut = blockIdx.x, mt = blockIdx.y;
  const int sm = (w & 3)*16, sn = (w >> 2)*32;

  // A: one 16B chunk per thread (64 rows x 4 chunks). k0 tiles (32-wide)
  // never straddle the [x | ctx | h] section boundaries (128, 640).
  const int am = tid >> 2, akc = (tid & 3) * 8;
  const int ab = mt*64 + am;
  auto stageA = [&](int buf, int k0){
    _Float16* dst = &As[buf][am*LDT + akc];
    if (k0 < EMBD){
      int tok = seq[ab*LSEQ + t];
      const float4* s = (const float4*)(emb + tok*EMBD + k0 + akc);
      *(v8h*)dst = cvt8(s[0], s[1]);
    } else {
      const _Float16* s = (k0 < EMBD + HIDN)
          ? (ctx16 + ab*HIDN + (k0 - EMBD) + akc)
          : (h16r  + ab*HIDN + (k0 - EMBD - HIDN) + akc);
      *(v8h*)dst = *(const v8h*)s;
    }
  };
  // B: 256 rows x 4 chunks = 4 chunks per thread, [N][K] in LDS
  auto stageB = [&](int buf, int k0){
#pragma unroll
    for (int r = 0; r < 4; ++r){
      int cid = tid + r*256;
      int n = cid >> 2, kc = (cid & 3) * 8;
      int g = n >> 6;
      size_t ncol = (size_t)(g*HIDN + ut*64 + (n & 63));
      *(v8h*)&Bs[buf][n*LDT + kc] = *(const v8h*)(Wcat + ncol*KCAT + k0 + kc);
      if (r == 0 && k0 + 32 < KCAT)
        __builtin_prefetch(Wcat + ncol*KCAT + k0 + 32 + kc, 0, 1);
    }
  };

  v8f acc[4][2] = {};
  stageA(0, 0); stageB(0, 0);
  const int NIT = KCAT/32;
  for (int kk = 0; kk < NIT; ++kk){
    __syncthreads();
    const int cur = kk & 1;
    if (kk + 1 < NIT){ stageA(cur ^ 1, (kk + 1)*32); stageB(cur ^ 1, (kk + 1)*32); }
    v16h a = frag_A(As[cur], lane, sm);
#pragma unroll
    for (int g = 0; g < 4; ++g){
#pragma unroll
      for (int j = 0; j < 2; ++j){
        v16h b = frag_B(Bs[cur], lane, g*64 + sn + j*16);
        acc[g][j] = WMMA(a, b, acc[g][j]);
      }
    }
  }
  // epilogue: LSTM cell update for this (row, hidden-unit) tile
  const int rb = mt*64 + sm + ((lane >> 4) << 3);
#pragma unroll
  for (int j = 0; j < 2; ++j){
    const int u = ut*64 + sn + j*16 + (lane & 15);
    const float bi = bih[u]        + bhh[u];
    const float bf = bih[HIDN+u]   + bhh[HIDN+u];
    const float bg = bih[2*HIDN+u] + bhh[2*HIDN+u];
    const float bo = bih[3*HIDN+u] + bhh[3*HIDN+u];
#pragma unroll
    for (int r = 0; r < 8; ++r){
      const int b = rb + r;
      float gi = acc[0][j][r] + bi;
      float gf = acc[1][j][r] + bf;
      float gg = acc[2][j][r] + bg;
      float go = acc[3][j][r] + bo;
      float c_old = cout[b*HIDN + u];
      float cn = sigf(gf)*c_old + sigf(gi)*tanhf(gg);
      float hn = sigf(go)*tanhf(cn);
      bool act = t < lens[b];
      if (act){ cout[b*HIDN + u] = cn; hout[b*HIDN + u] = hn; }
      h16w[b*HIDN + u] = act ? (_Float16)hn : h16r[b*HIDN + u]; // ping-pong
      out[((size_t)b*LSEQ + t)*HIDN + u] = act ? hn : 0.0f;
      hc16[b*1024 + u]        = (_Float16)hn;  // dec uses pre-mask h_new/c_new
      hc16[b*1024 + HIDN + u] = (_Float16)cn;
    }
  }
}

// -------- dec = [h_new|c_new] @ Wdp^T + bdp  (1024 x 512, K=1024) -----------
__global__ void __launch_bounds__(256) k_dec(const _Float16* __restrict__ hc16,
                                             const _Float16* __restrict__ Wdp16,
                                             const float* __restrict__ bdp,
                                             float* __restrict__ dec){
  __shared__ __align__(16) _Float16 As[2][64*LDT];
  __shared__ __align__(16) _Float16 Bs[2][64*LDT];
  const int tid = threadIdx.x, lane = tid & 31, w = tid >> 5;
  const int nt = blockIdx.x, mt = blockIdx.y;
  const int sm = (w & 3)*16, sn = (w >> 2)*32;
  const int m = tid >> 2, kc = (tid & 3) * 8;

  auto stage = [&](int buf, int k0){
    *(v8h*)&As[buf][m*LDT + kc] = *(const v8h*)(hc16  + (size_t)(mt*64 + m)*1024 + k0 + kc);
    *(v8h*)&Bs[buf][m*LDT + kc] = *(const v8h*)(Wdp16 + (size_t)(nt*64 + m)*1024 + k0 + kc);
  };

  v8f acc0 = {}, acc1 = {};
  stage(0, 0);
  const int NIT = 1024/32;
  for (int kk = 0; kk < NIT; ++kk){
    __syncthreads();
    const int cur = kk & 1;
    if (kk + 1 < NIT) stage(cur ^ 1, (kk + 1)*32);
    v16h a  = frag_A(As[cur], lane, sm);
    v16h b0 = frag_B(Bs[cur], lane, sn);
    v16h b1 = frag_B(Bs[cur], lane, sn + 16);
    acc0 = WMMA(a, b0, acc0);
    acc1 = WMMA(a, b1, acc1);
  }
  const int rb = mt*64 + sm + ((lane >> 4) << 3);
  const int cn = nt*64 + sn + (lane & 15);
#pragma unroll
  for (int r = 0; r < 8; ++r){
    dec[(rb + r)*HIDN + cn]      = acc0[r] + bdp[cn];
    dec[(rb + r)*HIDN + cn + 16] = acc1[r] + bdp[cn + 16];
  }
}

// -------- Bahdanau attention: scores->softmax->context (1 block / row) ------
__global__ void __launch_bounds__(256) k_attn(const _Float16* __restrict__ enc16,
                                              const float* __restrict__ dec,
                                              const float* __restrict__ vvec,
                                              const float* __restrict__ amask,
                                              const float* __restrict__ attn,
                                              const int* __restrict__ lens,
                                              _Float16* __restrict__ ctx16, int t)
{
  const int b = blockIdx.x;
  const int tid = threadIdx.x, lane = tid & 31, w = tid >> 5;
  __shared__ float sc[TKK];
  __shared__ float sa[TKK];
  const int h0 = lane * 16;                  // 512 / 32 lanes = 16 contiguous
  const float* dp = dec + (size_t)b*HIDN + h0;
  for (int tt = w; tt < TKK; tt += 8){
    const _Float16* e = enc16 + (size_t)(b*TKK + tt)*HIDN + h0;
    v8h e0 = *(const v8h*)(e);
    v8h e1 = *(const v8h*)(e + 8);
    float s = 0.0f;
#pragma unroll
    for (int i = 0; i < 16; ++i){
      float ev = (i < 8) ? (float)e0[i] : (float)e1[i - 8];
      s += tanhf(ev + dp[i]) * vvec[h0 + i];
    }
#pragma unroll
    for (int off = 16; off > 0; off >>= 1) s += __shfl_xor(s, off, 32);
    if (lane == 0) sc[tt] = s;
  }
  __syncthreads();
  float mx = -3.4e38f;
  for (int i = 0; i < TKK; ++i) mx = fmaxf(mx, sc[i]);
  if (tid < TKK) sa[tid] = __expf(sc[tid] - mx) * amask[b*TKK + tid];
  __syncthreads();
  float ssum = 0.0f;
  for (int i = 0; i < TKK; ++i) ssum += sa[i];
  const float inv = 1.0f / ssum;
  const bool act = t < lens[b];
  for (int h = tid; h < HIDN; h += 256){
    float acc = 0.0f;
    for (int tt = 0; tt < TKK; ++tt)
      acc += sa[tt] * attn[(size_t)(b*TKK + tt)*HIDN + h];
    if (act) ctx16[b*HIDN + h] = (_Float16)(acc * inv);
  }
}

extern "C" void kernel_launch(void* const* d_in, const int* in_sizes, int n_in,
                              void* d_out, int out_size, void* d_ws, size_t ws_size,
                              hipStream_t stream) {
  const int*   seq   = (const int*)d_in[0];
  const int*   lens  = (const int*)d_in[1];
  const float* h0    = (const float*)d_in[2];
  const float* c0    = (const float*)d_in[3];
  const float* attn  = (const float*)d_in[4];
  const float* amask = (const float*)d_in[5];
  const float* emb   = (const float*)d_in[6];
  const float* Wih   = (const float*)d_in[7];
  const float* Whh   = (const float*)d_in[8];
  const float* bih   = (const float*)d_in[9];
  const float* bhh   = (const float*)d_in[10];
  const float* Wh    = (const float*)d_in[11];
  const float* Wdp   = (const float*)d_in[12];
  const float* bdp   = (const float*)d_in[13];
  const float* vv    = (const float*)d_in[14];

  float* out  = (float*)d_out;                       // (B, L, HID)
  float* hout = out + (size_t)BTOT*LSEQ*HIDN;        // (1, B, HID)
  float* cout = hout + (size_t)BTOT*HIDN;            // (1, B, HID)

  char* p = (char*)d_ws;
  _Float16* Wcat16 = (_Float16*)p; p += (size_t)2048*KCAT*2;      // [N=2048][K=1152]
  _Float16* Wdp16  = (_Float16*)p; p += (size_t)HIDN*1024*2;      // [N=512][K=1024]
  _Float16* Wh16   = (_Float16*)p; p += (size_t)HIDN*HIDN*2;      // [N=512][K=512]
  _Float16* enc16  = (_Float16*)p; p += (size_t)BTOT*TKK*HIDN*2;  // enc_feat f16
  _Float16* h16a   = (_Float16*)p; p += (size_t)BTOT*HIDN*2;
  _Float16* h16b   = (_Float16*)p; p += (size_t)BTOT*HIDN*2;
  _Float16* ctx16  = (_Float16*)p; p += (size_t)BTOT*HIDN*2;
  _Float16* hc16   = (_Float16*)p; p += (size_t)BTOT*1024*2;
  float*    dec32  = (float*)p;    p += (size_t)BTOT*HIDN*4;
  if ((size_t)(p - (char*)d_ws) > ws_size) return;

  k_prep_wcat<<<(2048*KCAT + 255)/256, 256, 0, stream>>>(Wih, Whh, Wcat16);
  k_cvt<<<(HIDN*1024 + 255)/256, 256, 0, stream>>>(Wdp, Wdp16, HIDN*1024);
  k_cvt<<<(HIDN*HIDN + 255)/256, 256, 0, stream>>>(Wh, Wh16, HIDN*HIDN);
  k_init<<<(BTOT*HIDN + 255)/256, 256, 0, stream>>>(h0, c0, hout, cout, h16a, ctx16);
  k_encfeat<<<dim3(HIDN/64, (BTOT*TKK)/64), 256, 0, stream>>>(attn, Wh16, enc16);

  for (int t = 0; t < LSEQ; ++t){
    _Float16* hr = (t & 1) ? h16b : h16a;
    _Float16* hw = (t & 1) ? h16a : h16b;
    k_lstm<<<dim3(HIDN/64, BTOT/64), 256, 0, stream>>>(seq, lens, emb, Wcat16, ctx16,
                                                       hr, hw, bih, bhh,
                                                       cout, hout, out, hc16, t);
    k_dec<<<dim3(HIDN/64, BTOT/64), 256, 0, stream>>>(hc16, Wdp16, bdp, dec32);
    k_attn<<<BTOT, 256, 0, stream>>>(enc16, dec32, vv, amask, attn, lens, ctx16, t);
  }
}
